// decoder_77051713290943
// MI455X (gfx1250) — compile-verified, matrix operand order checked by reading, not run on hardware
//
#include <hip/hip_runtime.h>
#include <cstdint>
#include <cstddef>

// ---------------------------------------------------------------------------
// Persistent bf16-WMMA LSTM decoder for gfx1250 (MI455X).
//   B=1024 rows -> 32 workgroups x 32 rows, no inter-WG communication.
//   H=300 padded: N -> 304 (19 tiles of 16), K -> 320 (10 chunks of 32).
//   Weights pre-packed to bf16 WMMA B-fragment layout in d_ws (streams from L2):
//     gate weights grouped so the 8 fragments of one (ntile,kchunk) form one
//     contiguous 8KB block -> single running pointer + immediate offsets.
//   c state lives in VGPRs for all timesteps; x/h live in LDS as bf16 tiles.
// ---------------------------------------------------------------------------

typedef unsigned short u16;
typedef u16    u16x8   __attribute__((ext_vector_type(8)));
typedef u16    u16x16  __attribute__((ext_vector_type(16)));
typedef __bf16 bf16x16 __attribute__((ext_vector_type(16)));
typedef float  f32x8   __attribute__((ext_vector_type(8)));

#define HDIM   300
#define KP     320           // padded K (multiple of 32)
#define NTILE  19            // 304/16 N-tiles per gate (and for W_out)
#define KCH    10            // 320/32 K-chunks
#define NWAVE  5
#define NTHREADS 160         // 5 wave32
#define MAXT   4             // max owned tiles per wave (19/5 -> 4)
#define BROWS  32            // batch rows per workgroup
#define NBLK   32            // 1024/32
#define BATCH  1024

#define FRAG_U16   512                       // one 32x16 bf16 fragment = 1KB
#define GBLK_U16   (8 * FRAG_U16)            // 8 matrices per (nt,kc) = 8KB block

// ws region element counts
#define GFRAG_ELEMS (NTILE * KCH * GBLK_U16) // i,f,g,o x {W_ih,W_hh} fragments
#define OFRAG_ELEMS (NTILE * KCH * FRAG_U16) // W_out fragments
#define GB_ELEMS    (4 * 304)                // fused gate bias, padded
#define OB_ELEMS    (304)                    // b_out, padded

__device__ __forceinline__ u16 f2bf(float f) {           // round-to-nearest-even
  unsigned u = __float_as_uint(f);
  u = (u + 0x7FFFu + ((u >> 16) & 1u)) >> 16;
  return (u16)u;
}
__device__ __forceinline__ float sigf(float x)  { return 1.0f / (1.0f + __expf(-x)); }
__device__ __forceinline__ float tanhf_(float x){ return 1.0f - 2.0f / (__expf(2.0f * x) + 1.0f); }

__device__ __forceinline__ f32x8 wmma_bf16(u16x16 a, u16x16 b, f32x8 c) {
  return __builtin_amdgcn_wmma_f32_16x16x32_bf16(
      false, __builtin_bit_cast(bf16x16, a),
      false, __builtin_bit_cast(bf16x16, b),
      (short)0, c, false, false);
}

// A-fragment (16x32 bf16) from an LDS [32 x KP] bf16 tile.
// ISA layout: lane L<16 -> M=L, K = kc*32 + {0..7, 16..23}; lane L+16 -> K = {8..15, 24..31}.
__device__ __forceinline__ u16x16 lds_afrag(const u16* buf, int mbase, int kc, int lane) {
  int row = mbase + (lane & 15);
  int kb  = kc * 32 + ((lane >> 4) << 3);
  const u16* p = buf + row * KP + kb;
  u16x8 lo = *(const u16x8*)(p);        // ds_load_b128 (immediate offsets fold)
  u16x8 hi = *(const u16x8*)(p + 16);   // ds_load_b128
  return __builtin_shufflevector(lo, hi, 0,1,2,3,4,5,6,7,8,9,10,11,12,13,14,15);
}

// ---------------------------------------------------------------------------
// Pre-pack: f32 weights -> bf16 B-fragment-major layout + fused biases.
// B[k][n] = W[n][k]; fragment is 32 lanes x 16 bf16 (lane-major, 1KB).
// Lane mirror of the A layout: lane L<16 -> N=L, K={0..7,16..23}; L+16 -> K={8..15,24..31}.
// Gate frag element index: (((nt*KCH + kc)*8 + m) * 512) + lane*16 + e,  m = gate*2+hh.
// ---------------------------------------------------------------------------
__global__ void pack_kernel(const float* __restrict__ Wih, const float* __restrict__ Whh,
                            const float* __restrict__ Wout,
                            const float* __restrict__ bih, const float* __restrict__ bhh,
                            const float* __restrict__ bout,
                            u16* __restrict__ gw, u16* __restrict__ ow,
                            float* __restrict__ gb, float* __restrict__ ob) {
  int id = blockIdx.x * blockDim.x + threadIdx.x;
  const int R0 = GFRAG_ELEMS;
  const int R1 = R0 + OFRAG_ELEMS;
  const int R2 = R1 + GB_ELEMS;
  const int R3 = R2 + OB_ELEMS;
  if (id < R0) {
    int fragid = id >> 9, r = id & 511;
    int lane = r >> 4, e = r & 15;
    int m   = fragid & 7;
    int blk = fragid >> 3;
    int nt = blk / KCH, kc = blk % KCH;
    int gate = m >> 1, hh = m & 1;
    int n = nt * 16 + (lane & 15);
    int k = kc * 32 + ((lane >> 4) << 3) + e + ((e >= 8) ? 8 : 0);
    float v = 0.f;
    if (n < HDIM && k < HDIM) {
      const float* src = hh ? Whh : Wih;
      v = src[(gate * HDIM + n) * HDIM + k];
    }
    gw[id] = f2bf(v);
  } else if (id < R1) {
    int idx = id - R0;
    int fragid = idx >> 9, r = idx & 511;
    int lane = r >> 4, e = r & 15;
    int nt = fragid / KCH, kc = fragid % KCH;
    int n = nt * 16 + (lane & 15);
    int k = kc * 32 + ((lane >> 4) << 3) + e + ((e >= 8) ? 8 : 0);
    float v = 0.f;
    if (n < HDIM && k < HDIM) v = Wout[n * HDIM + k];
    ow[idx] = f2bf(v);
  } else if (id < R2) {
    int idx = id - R1;
    int g = idx / 304, j = idx % 304;
    gb[idx] = (j < HDIM) ? (bih[g * HDIM + j] + bhh[g * HDIM + j]) : 0.f;
  } else if (id < R3) {
    int j = id - R2;
    ob[j] = (j < HDIM) ? bout[j] : 0.f;
  }
}

// ---------------------------------------------------------------------------
// Persistent decoder kernel.
// ---------------------------------------------------------------------------
__global__ __launch_bounds__(NTHREADS)
void lstm_persist(const float* __restrict__ h0, const float* __restrict__ c0,
                  const float* __restrict__ start,
                  const u16* __restrict__ gw, const u16* __restrict__ ow,
                  const float* __restrict__ gb, const float* __restrict__ ob,
                  const int* __restrict__ oplen, float* __restrict__ out) {
  __shared__ u16 sX[BROWS * KP];         // current input x_t (bf16)
  __shared__ u16 sH[2][BROWS * KP];      // ping-pong hidden state (bf16)

  const int tid  = threadIdx.x;
  const int lane = tid & 31;
  const int wave = tid >> 5;
  const int row0 = blockIdx.x * BROWS;
  const int T    = oplen[0];

  // ---- init LDS tiles from start / h0 (f32 -> bf16, zero the K padding)
  for (int i = tid; i < BROWS * KP; i += NTHREADS) {
    int r = i / KP, c = i - r * KP;
    u16 xv = 0, hv = 0;
    if (c < HDIM) {
      xv = f2bf(start[(size_t)(row0 + r) * HDIM + c]);
      hv = f2bf(h0[(size_t)(row0 + r) * HDIM + c]);
    }
    sX[i] = xv; sH[0][i] = hv; sH[1][i] = 0;
  }

  const int colJ   = lane & 15;          // C/D: N = lane%16
  const int rowOff = (lane >> 4) << 3;   // C/D: M = r + 8*(lane/16)

  // ---- per-wave persistent state: c in registers, biases in registers
  float creg[MAXT][2][8];
  float bI[MAXT], bF[MAXT], bG[MAXT], bO[MAXT], bY[MAXT];
#pragma unroll
  for (int s = 0; s < MAXT; ++s) {
    bI[s] = bF[s] = bG[s] = bO[s] = bY[s] = 0.f;
#pragma unroll
    for (int mh = 0; mh < 2; ++mh)
#pragma unroll
      for (int r = 0; r < 8; ++r) creg[s][mh][r] = 0.f;
    int nt = wave + s * NWAVE;
    if (nt < NTILE) {
      int j = nt * 16 + colJ;            // j <= 303
      bI[s] = gb[0 * 304 + j]; bF[s] = gb[1 * 304 + j];
      bG[s] = gb[2 * 304 + j]; bO[s] = gb[3 * 304 + j];
      bY[s] = ob[j];
      if (j < HDIM) {
#pragma unroll
        for (int mh = 0; mh < 2; ++mh)
#pragma unroll
          for (int r = 0; r < 8; ++r) {
            int row = row0 + mh * 16 + r + rowOff;
            creg[s][mh][r] = c0[(size_t)row * HDIM + j];
          }
      }
    }
  }
  __syncthreads();

  for (int t = 0; t < T; ++t) {
    const u16* Hin = sH[t & 1];
    u16*       Hout = sH[1 - (t & 1)];

    // ===== phase 1: gates = x@W^T + h@U^T + b ; LSTM update; publish h_new =====
#pragma unroll
    for (int s = 0; s < MAXT; ++s) {
      int nt = wave + s * NWAVE;
      if (nt >= NTILE) continue;
      f32x8 acc[4][2];
#pragma unroll
      for (int g = 0; g < 4; ++g) {
        float bb = (g == 0) ? bI[s] : (g == 1) ? bF[s] : (g == 2) ? bG[s] : bO[s];
#pragma unroll
        for (int mh = 0; mh < 2; ++mh)
#pragma unroll
          for (int e = 0; e < 8; ++e) acc[g][mh][e] = bb;
      }
      // single running pointer per tile: (nt,kc) block of 8 contiguous 1KB frags
      const u16* pB = gw + (size_t)nt * KCH * GBLK_U16 + lane * 16;
#pragma unroll 1
      for (int kc = 0; kc < KCH; ++kc) {
        u16x16 ax0 = lds_afrag(sX,  0, kc, lane);
        u16x16 ax1 = lds_afrag(sX, 16, kc, lane);
        u16x16 ah0 = lds_afrag(Hin,  0, kc, lane);
        u16x16 ah1 = lds_afrag(Hin, 16, kc, lane);
        if (kc + 1 < KCH) {   // prefetch next 8KB block into near cache (WGP)
#pragma unroll
          for (int m = 0; m < 8; ++m)
            __builtin_prefetch(pB + GBLK_U16 + m * FRAG_U16, 0, 3);
        }
#pragma unroll
        for (int g = 0; g < 4; ++g) {
          // immediate-offset loads off the running pointer (folds into ioffset)
          u16x16 bw = *(const u16x16*)(pB + (2 * g + 0) * FRAG_U16);
          u16x16 bu = *(const u16x16*)(pB + (2 * g + 1) * FRAG_U16);
          acc[g][0] = wmma_bf16(ax0, bw, acc[g][0]);
          acc[g][1] = wmma_bf16(ax1, bw, acc[g][1]);
          acc[g][0] = wmma_bf16(ah0, bu, acc[g][0]);
          acc[g][1] = wmma_bf16(ah1, bu, acc[g][1]);
        }
        pB += GBLK_U16;
      }
      // elementwise LSTM update, all in registers; h_new -> Hout (bf16)
      int col = nt * 16 + colJ;
#pragma unroll
      for (int mh = 0; mh < 2; ++mh)
#pragma unroll
        for (int r = 0; r < 8; ++r) {
          float iv = sigf(acc[0][mh][r]);
          float fv = sigf(acc[1][mh][r]);
          float gv = tanhf_(acc[2][mh][r]);
          float ov = sigf(acc[3][mh][r]);
          float cn = fv * creg[s][mh][r] + iv * gv;
          creg[s][mh][r] = cn;
          float hn = ov * tanhf_(cn);
          Hout[(mh * 16 + r + rowOff) * KP + col] = f2bf(hn);
        }
    }
    __syncthreads();   // h_new fully published; all phase-1 reads of sX done

    // ===== phase 3: out = h_new @ W_out^T + b_out; store + feed back as x =====
#pragma unroll
    for (int s = 0; s < MAXT; ++s) {
      int nt = wave + s * NWAVE;
      if (nt >= NTILE) continue;
      f32x8 oa[2];
#pragma unroll
      for (int mh = 0; mh < 2; ++mh)
#pragma unroll
        for (int e = 0; e < 8; ++e) oa[mh][e] = bY[s];
      const u16* pO = ow + (size_t)nt * KCH * FRAG_U16 + lane * 16;
#pragma unroll 1
      for (int kc = 0; kc < KCH; ++kc) {
        u16x16 a0 = lds_afrag(Hout,  0, kc, lane);
        u16x16 a1 = lds_afrag(Hout, 16, kc, lane);
        u16x16 b  = *(const u16x16*)(pO);
        if (kc + 1 < KCH)
          __builtin_prefetch(pO + FRAG_U16, 0, 3);
        oa[0] = wmma_bf16(a0, b, oa[0]);
        oa[1] = wmma_bf16(a1, b, oa[1]);
        pO += FRAG_U16;
      }
      int col = nt * 16 + colJ;
#pragma unroll
      for (int mh = 0; mh < 2; ++mh)
#pragma unroll
        for (int r = 0; r < 8; ++r) {
          float v = oa[mh][r];
          int rl = mh * 16 + r + rowOff;
          sX[rl * KP + col] = f2bf(v);
          if (col < HDIM)
            out[((size_t)t * BATCH + row0 + rl) * HDIM + col] = v;
        }
    }
    __syncthreads();   // sX fully rewritten before next step's gates GEMM
  }
}

// ---------------------------------------------------------------------------
extern "C" void kernel_launch(void* const* d_in, const int* in_sizes, int n_in,
                              void* d_out, int out_size, void* d_ws, size_t ws_size,
                              hipStream_t stream) {
  (void)in_sizes; (void)n_in; (void)out_size;
  const float* h0    = (const float*)d_in[0];
  const float* c0    = (const float*)d_in[1];
  const float* start = (const float*)d_in[2];
  const float* W_ih  = (const float*)d_in[3];
  const float* W_hh  = (const float*)d_in[4];
  const float* b_ih  = (const float*)d_in[5];
  const float* b_hh  = (const float*)d_in[6];
  const float* W_out = (const float*)d_in[7];
  const float* b_out = (const float*)d_in[8];
  const int*   oplen = (const int*)d_in[9];
  float* outp = (float*)d_out;

  // workspace partition (~1.76 MB needed)
  u16*   gw = (u16*)d_ws;
  u16*   ow = gw + GFRAG_ELEMS;
  float* gb = (float*)(ow + OFRAG_ELEMS);   // byte offset 1,751,040 (16B aligned)
  float* ob = gb + GB_ELEMS;
  (void)ws_size;

  const int total = GFRAG_ELEMS + OFRAG_ELEMS + GB_ELEMS + OB_ELEMS;
  int pblk = (total + 255) / 256;
  pack_kernel<<<pblk, 256, 0, stream>>>(W_ih, W_hh, W_out, b_ih, b_hh, b_out,
                                        gw, ow, gb, ob);
  lstm_persist<<<NBLK, NTHREADS, 0, stream>>>(h0, c0, start, gw, ow, gb, ob,
                                              oplen, outp);
}
